// RNN_21277267984515
// MI455X (gfx1250) — compile-verified, matrix operand order checked by reading, not run on hardware
//
#include <hip/hip_runtime.h>
#include <hip/hip_bf16.h>
#include <math.h>

#define NB 64      // batch
#define TT 512     // timesteps
#define DD 1024    // input dim
#define HH 1024    // hidden dim

#define LDST 40    // LDS row stride in bf16 elements (80B: padded, 16B-aligned)

typedef __bf16 bf16_t;
typedef __attribute__((ext_vector_type(16))) __bf16 v16bf;
typedef __attribute__((ext_vector_type(8)))  __bf16 v8bf;
typedef __attribute__((ext_vector_type(8)))  float  v8f;
typedef __attribute__((ext_vector_type(4)))  int    v4i;

// ---------------------------------------------------------------- helpers

__device__ __forceinline__ bf16_t f2bf(float f) {
    unsigned u = __builtin_bit_cast(unsigned, f);
    unsigned r = u + 0x7FFFu + ((u >> 16) & 1u);   // round-to-nearest-even
    unsigned short h = (unsigned short)(r >> 16);
    return __builtin_bit_cast(bf16_t, h);
}

// 16-byte global -> LDS copy; prefer CDNA5 async path (ASYNCcnt-tracked).
#if __has_builtin(__builtin_amdgcn_global_load_async_to_lds_b128)
#define HAVE_ASYNC_LDS 1
__device__ __forceinline__ void async_cp16(bf16_t* lds_dst, const bf16_t* gsrc) {
    __builtin_amdgcn_global_load_async_to_lds_b128(
        (__attribute__((address_space(1))) v4i*)(v4i*)(void*)gsrc,
        (__attribute__((address_space(3))) v4i*)(v4i*)(void*)lds_dst,
        0, 0);
}
#else
#define HAVE_ASYNC_LDS 0
__device__ __forceinline__ void async_cp16(bf16_t* lds_dst, const bf16_t* gsrc) {
    *(v8bf*)lds_dst = *(const v8bf*)gsrc;   // sync fallback: load + ds_store
}
#endif

__device__ __forceinline__ void wait_async() {
#if __has_builtin(__builtin_amdgcn_s_wait_asynccnt)
    __builtin_amdgcn_s_wait_asynccnt(0);
#else
    asm volatile("s_wait_asynccnt 0x0" ::: "memory");
#endif
}

// Load one 16x32 bf16 fragment from an LDS tile (rows x LDST layout).
// lanes 0-15: row = lane,    K = {0..7, 16..23}
// lanes16-31: row = lane-16, K = {8..15, 24..31}
__device__ __forceinline__ v16bf lds_frag(const bf16_t* tile, int lane) {
    const bf16_t* p = tile + (lane & 15) * LDST + ((lane & 16) ? 8 : 0);
    v8bf lo = *(const v8bf*)(p);        // ds_load_b128
    v8bf hi = *(const v8bf*)(p + 16);   // ds_load_b128
    v16bf r;
#pragma unroll
    for (int i = 0; i < 8; ++i) { r[i] = lo[i]; r[i + 8] = hi[i]; }
    return r;
}

// ---------------------------------------------------------------- prep kernels

__global__ void cvt_bf16_kernel(const float* __restrict__ src,
                                bf16_t* __restrict__ dst, size_t n) {
    size_t i = (size_t)blockIdx.x * blockDim.x + threadIdx.x;
    if (i < n) dst[i] = f2bf(src[i]);
}

// dst[c*rows + r] = src[r*cols + c]   (fp32 -> bf16 transpose)
__global__ void transpose_bf16_kernel(const float* __restrict__ src,
                                      bf16_t* __restrict__ dst,
                                      int rows, int cols) {
    size_t i = (size_t)blockIdx.x * blockDim.x + threadIdx.x;
    size_t n = (size_t)rows * cols;
    if (i < n) {
        int c = (int)(i / rows);
        int r = (int)(i % rows);
        dst[i] = f2bf(src[(size_t)r * cols + c]);
    }
}

// ---------------------------------------------------------------- phase 1 GEMM
// out[m, h] = xb[m, :] . WxT[h, :] + b[h]     (M = N*T = 32768)
// Block: 256 thr / 8 waves; macro-tile 128M x 64N; LDS double-buffered tiles.

__global__ void __launch_bounds__(256)
xw_gemm_kernel(const bf16_t* __restrict__ xb,
               const bf16_t* __restrict__ WxT,
               const float*  __restrict__ bias,
               float* __restrict__ out) {
    __shared__ __align__(16) bf16_t Abuf[2][128][LDST];  // 20 KB
    __shared__ __align__(16) bf16_t Bbuf[2][64][LDST];   // 10 KB

    const int tid  = threadIdx.x;
    const int wave = tid >> 5;
    const int lane = tid & 31;
    const int m0 = (blockIdx.x >> 4) * 128;   // 256 M-blocks
    const int n0 = (blockIdx.x & 15) * 64;    // 16  N-blocks

    const bf16_t* Ab = xb  + (size_t)m0 * DD;
    const bf16_t* Bb = WxT + (size_t)n0 * DD;

    // A tile: 128 rows x 32k = 512 16B-chunks (2/thread); B: 64 rows = 256 (1/thread)
    auto stage = [&](int buf, int kb) {
#pragma unroll
        for (int s = 0; s < 2; ++s) {
            int idx = tid + s * 256;
            int row = idx >> 2, ch = idx & 3;
            async_cp16(&Abuf[buf][row][ch * 8],
                       Ab + (size_t)row * DD + kb * 32 + ch * 8);
        }
        { int row = tid >> 2, ch = tid & 3;
          async_cp16(&Bbuf[buf][row][ch * 8],
                     Bb + (size_t)row * DD + kb * 32 + ch * 8); }
    };

    stage(0, 0);
    wait_async();
    __syncthreads();

    v8f acc[4] = {};
    for (int kb = 0; kb < DD / 32; ++kb) {
        const int cur = kb & 1, nxt = cur ^ 1;
        if (kb + 1 < DD / 32) stage(nxt, kb + 1);   // overlap copy with compute
        v16bf a = lds_frag(&Abuf[cur][wave * 16][0], lane);
#pragma unroll
        for (int j = 0; j < 4; ++j) {
            v16bf b = lds_frag(&Bbuf[cur][j * 16][0], lane);
            acc[j] = __builtin_amdgcn_wmma_f32_16x16x32_bf16(
                false, a, false, b, (short)0, acc[j], false, false);
        }
        wait_async();
        __syncthreads();
    }

    const int col_in = lane & 15;
    const int rowadd = (lane & 16) ? 8 : 0;
    const int mrow0  = m0 + wave * 16;
#pragma unroll
    for (int j = 0; j < 4; ++j) {
        int col = n0 + j * 16 + col_in;
        float bv = bias[col];
#pragma unroll
        for (int r = 0; r < 8; ++r)
            out[(size_t)(mrow0 + rowadd + r) * HH + col] = acc[j][r] + bv;
    }
}

// ---------------------------------------------------------------- step kernel
// h_next = tanh(h_prev @ Wh + xw_t); xw_t staged in-place in out[:, t, :].
// Block covers all M (64) x 64 N-cols -> 16 blocks; 8 waves = 4M x 2N tiles.

__global__ void __launch_bounds__(256)
rnn_step_kernel(const bf16_t* __restrict__ hprev,
                const bf16_t* __restrict__ WhT,
                bf16_t* __restrict__ hnext,
                float* __restrict__ out, int t) {
    __shared__ __align__(16) bf16_t Abuf[2][64][LDST];   // h tile, 10 KB
    __shared__ __align__(16) bf16_t Bbuf[2][64][LDST];   // WhT tile, 10 KB

    const int tid  = threadIdx.x;
    const int wave = tid >> 5;
    const int lane = tid & 31;
    const int n0   = blockIdx.x * 64;

    const bf16_t* Bb = WhT + (size_t)n0 * HH;

    auto stage = [&](int buf, int kb) {
        int row = tid >> 2, ch = tid & 3;
        async_cp16(&Abuf[buf][row][ch * 8],
                   hprev + (size_t)row * HH + kb * 32 + ch * 8);
        async_cp16(&Bbuf[buf][row][ch * 8],
                   Bb + (size_t)row * HH + kb * 32 + ch * 8);
    };

    stage(0, 0);
    wait_async();
    __syncthreads();

    const int mw = (wave & 3) * 16;     // wave M-tile (global batch rows)
    const int nw = (wave >> 2) * 32;    // wave N-group within block

    v8f acc[2] = {};
    for (int kb = 0; kb < HH / 32; ++kb) {
        const int cur = kb & 1, nxt = cur ^ 1;
        if (kb + 1 < HH / 32) stage(nxt, kb + 1);
        v16bf a = lds_frag(&Abuf[cur][mw][0], lane);
#pragma unroll
        for (int j = 0; j < 2; ++j) {
            v16bf b = lds_frag(&Bbuf[cur][nw + j * 16][0], lane);
            acc[j] = __builtin_amdgcn_wmma_f32_16x16x32_bf16(
                false, a, false, b, (short)0, acc[j], false, false);
        }
        wait_async();
        __syncthreads();
    }

    const int col_in = lane & 15;
    const int rowadd = (lane & 16) ? 8 : 0;
#pragma unroll
    for (int j = 0; j < 2; ++j) {
        int col = n0 + nw + j * 16 + col_in;
#pragma unroll
        for (int r = 0; r < 8; ++r) {
            int m = mw + rowadd + r;
            size_t oidx = ((size_t)m * TT + t) * HH + col;   // (n, t, h)
            float v = tanhf(acc[j][r] + out[oidx]);
            out[oidx] = v;
            hnext[(size_t)m * HH + col] = f2bf(v);
        }
    }
}

// ---------------------------------------------------------------- launch

extern "C" void kernel_launch(void* const* d_in, const int* in_sizes, int n_in,
                              void* d_out, int out_size, void* d_ws, size_t ws_size,
                              hipStream_t stream) {
    const float* x  = (const float*)d_in[0];   // (N, T, D)
    const float* h0 = (const float*)d_in[1];   // (N, H)
    const float* Wx = (const float*)d_in[2];   // (D, H)
    const float* Wh = (const float*)d_in[3];   // (H, H)
    const float* b  = (const float*)d_in[4];   // (H,)
    float* out = (float*)d_out;                // (N, T, H)

    char* ws = (char*)d_ws;
    size_t off = 0;
    bf16_t* xb  = (bf16_t*)(ws + off); off += (size_t)NB * TT * DD * 2;  // 64 MB
    bf16_t* WxT = (bf16_t*)(ws + off); off += (size_t)HH * DD * 2;       //  2 MB
    bf16_t* WhT = (bf16_t*)(ws + off); off += (size_t)HH * HH * 2;       //  2 MB
    bf16_t* hb0 = (bf16_t*)(ws + off); off += (size_t)NB * HH * 2;
    bf16_t* hb1 = (bf16_t*)(ws + off); off += (size_t)NB * HH * 2;

    // --- prep: fp32 -> bf16 (x, h0), transposed bf16 weights
    {
        size_t n = (size_t)NB * TT * DD;
        cvt_bf16_kernel<<<(unsigned)((n + 255) / 256), 256, 0, stream>>>(x, xb, n);
    }
    {
        size_t n = (size_t)NB * HH;
        cvt_bf16_kernel<<<(unsigned)((n + 255) / 256), 256, 0, stream>>>(h0, hb0, n);
    }
    {
        size_t n = (size_t)DD * HH;
        transpose_bf16_kernel<<<(unsigned)((n + 255) / 256), 256, 0, stream>>>(Wx, WxT, DD, HH);
        transpose_bf16_kernel<<<(unsigned)((n + 255) / 256), 256, 0, stream>>>(Wh, WhT, HH, HH);
    }

    // --- phase 1: out = x @ Wx + b   (staged into d_out)
    {
        int blocks = ((NB * TT) / 128) * (HH / 64);   // 256 * 16 = 4096
        xw_gemm_kernel<<<blocks, 256, 0, stream>>>(xb, WxT, b, out);
    }

    // --- phase 2: 512 sequential steps (WhT stays hot in L2)
    for (int t = 0; t < TT; ++t) {
        bf16_t* src = (t & 1) ? hb1 : hb0;
        bf16_t* dst = (t & 1) ? hb0 : hb1;
        rnn_step_kernel<<<HH / 64, 256, 0, stream>>>(src, WhT, dst, out, t);
    }
    (void)in_sizes; (void)n_in; (void)out_size; (void)ws_size;
}
